// SerriformBlock_41120016891974
// MI455X (gfx1250) — matrix operand order repaired and checked by reading, not compiled, not run on hardware
//
#include <hip/hip_runtime.h>
#include <cmath>

#define DD 512
#define MM 128
#define EE 4
#define KCONV 5
#define DIL 2
#define HH 128
#define BBATCH 4
#define LSEQ 8192
#define NT (BBATCH * LSEQ)   // 32768 tokens

typedef __attribute__((ext_vector_type(16))) __bf16 v16bf;
typedef __attribute__((ext_vector_type(8)))  __bf16 v8bf;
typedef __attribute__((ext_vector_type(8)))  float  v8f;

constexpr int LDSP = 72;   // halves per LDS row (144B: 16B-aligned, bank-step 36 -> conflict-free)
constexpr int BK   = 64;   // K elements staged per iteration (2 WMMA k-steps)

__device__ __forceinline__ float fast_rcp(float x) { return __builtin_amdgcn_rcpf(x); }
__device__ __forceinline__ float sigm(float x) { return fast_rcp(1.f + __expf(-x)); }

// A fragment (16x32 bf16): lane holds M=lane&15; halves 0..7 -> K=ks+(lane>>4)*8..,
// halves 8..15 -> K=ks+16+(lane>>4)*8 (per ISA 16-bit A layout table)
__device__ __forceinline__ v16bf fragA(const __bf16* s, int mOff, int lane, int ks) {
  const __bf16* p = s + (mOff + (lane & 15)) * LDSP + ks + ((lane >> 4) << 3);
  v8bf lo = *(const v8bf*)p;
  v8bf hi = *(const v8bf*)(p + 16);
  return __builtin_shufflevector(lo, hi, 0,1,2,3,4,5,6,7,8,9,10,11,12,13,14,15);
}
// B fragment (32x16 bf16): lane holds N=lane&15; halves 0..7 -> K=ks+(lane>>4)*8..,
// halves 8..15 -> K=ks+16+(lane>>4)*8
__device__ __forceinline__ v16bf fragB(const __bf16* s, int nOff, int lane, int ks) {
  const __bf16* p = s + (nOff + (lane & 15)) * LDSP + ks + ((lane >> 4) << 3);
  v8bf lo = *(const v8bf*)p;
  v8bf hi = *(const v8bf*)(p + 16);
  return __builtin_shufflevector(lo, hi, 0,1,2,3,4,5,6,7,8,9,10,11,12,13,14,15);
}

// Epilogues: 0=plain f32 (+bias) 1=silu f32 2=gate (+bias+xn+inplace xconv, also bf16 h)
// 3=expert accumulate (+= g*silu, also bf16 copy) 4=gelu -> bf16 only 5=final (+bias+h2+rs*resid)
template<int EPI>
__global__ __launch_bounds__(128)
void gemm_wmma(const __bf16* __restrict__ A1, const __bf16* __restrict__ A2,
               int K1, int Ktot,
               const __bf16* __restrict__ W, const float* __restrict__ bias,
               float* __restrict__ out, __bf16* __restrict__ outbf, int Ncols,
               const float* __restrict__ add1,
               const float* __restrict__ gates, const int* __restrict__ gidx,
               int expert,
               const float* __restrict__ resid, const float* __restrict__ rscale)
{
  __shared__ __bf16 sA[64 * LDSP];
  __shared__ __bf16 sB[64 * LDSP];
  const int tid  = threadIdx.x;
  const int lane = tid & 31;
  const int wid  = tid >> 5;
  const int mOff = (wid >> 1) * 32;
  const int nOff = (wid & 1) * 32;
  const int rowBase = blockIdx.y * 64;
  const int colBase = blockIdx.x * 64;
  const int K2 = Ktot - K1;

  v8f acc00{}, acc01{}, acc10{}, acc11{};

  const int srow = tid >> 1;         // staged row 0..63
  const int scol = (tid & 1) << 5;   // 0 or 32 bf16 elems (64B per thread)

  auto gA = [&](int kb) -> const __bf16* {
    return (kb < K1) ? A1 + (size_t)(rowBase + srow) * K1 + kb + scol
                     : A2 + (size_t)(rowBase + srow) * K2 + (kb - K1) + scol;
  };
  const __bf16* gB = W + (size_t)(colBase + srow) * Ktot + scol;
  __bf16* const dA = sA + srow * LDSP + scol;
  __bf16* const dB = sB + srow * LDSP + scol;

  // warm L1/L2 for the first tiles (register-free latency hiding)
  __builtin_prefetch(gA(0), 0, 1);
  __builtin_prefetch(gB, 0, 1);

  for (int kb = 0; kb < Ktot; kb += BK) {
    __syncthreads();                 // previous iteration's readers done
    { // stage current tile: pure bf16 copies, staging regs die immediately
      const uint4* pa = (const uint4*)gA(kb);
      const uint4* pb = (const uint4*)(gB + kb);
      uint4 a0 = pa[0], a1 = pa[1], a2 = pa[2], a3 = pa[3];
      uint4 b0 = pb[0], b1 = pb[1], b2 = pb[2], b3 = pb[3];
      if (kb + BK < Ktot) {          // prefetch next tile while this one lands
        __builtin_prefetch(gA(kb + BK), 0, 1);
        __builtin_prefetch(gB + kb + BK, 0, 1);
      }
      uint4* qa = (uint4*)dA;  qa[0] = a0; qa[1] = a1; qa[2] = a2; qa[3] = a3;
      uint4* qb = (uint4*)dB;  qb[0] = b0; qb[1] = b1; qb[2] = b2; qb[3] = b3;
    }
    __syncthreads();                 // tile visible to all waves
    #pragma unroll
    for (int ks = 0; ks < BK; ks += 32) {
      v16bf a0 = fragA(sA, mOff,      lane, ks);
      v16bf a1 = fragA(sA, mOff + 16, lane, ks);
      v16bf b0 = fragB(sB, nOff,      lane, ks);
      v16bf b1 = fragB(sB, nOff + 16, lane, ks);
      acc00 = __builtin_amdgcn_wmma_f32_16x16x32_bf16(false, a0, false, b0, (short)0, acc00, false, false);
      acc01 = __builtin_amdgcn_wmma_f32_16x16x32_bf16(false, a0, false, b1, (short)0, acc01, false, false);
      acc10 = __builtin_amdgcn_wmma_f32_16x16x32_bf16(false, a1, false, b0, (short)0, acc10, false, false);
      acc11 = __builtin_amdgcn_wmma_f32_16x16x32_bf16(false, a1, false, b1, (short)0, acc11, false, false);
    }
  }

  const float rs = (EPI == 5) ? rscale[0] : 0.f;
  #pragma unroll
  for (int mt = 0; mt < 2; ++mt) {
    #pragma unroll
    for (int nt = 0; nt < 2; ++nt) {
      v8f acc = (mt == 0) ? (nt == 0 ? acc00 : acc01) : (nt == 0 ? acc10 : acc11);
      #pragma unroll
      for (int r = 0; r < 8; ++r) {
        // C/D layout: lane -> N, VGPR r -> M = r + 8*(lane>=16)
        int row = rowBase + mOff + mt * 16 + ((lane >> 4) << 3) + r;
        int col = colBase + nOff + nt * 16 + (lane & 15);
        size_t idx = (size_t)row * Ncols + col;
        float v = acc[r] + bias[col];
        if constexpr (EPI == 0) {
          out[idx] = v;
        } else if constexpr (EPI == 1) {
          out[idx] = v * sigm(v);
        } else if constexpr (EPI == 2) {
          float hv = v + add1[idx] + out[idx];   // h = gemm+b + xn + x_conv(inplace)
          out[idx] = hv;
          outbf[idx] = (__bf16)hv;
        } else if constexpr (EPI == 3) {
          float g = 0.f;
          int i0 = gidx[row * 2], i1 = gidx[row * 2 + 1];
          if (i0 == expert)      g = gates[row * 2];
          else if (i1 == expert) g = gates[row * 2 + 1];
          float nv = out[idx] + g * (v * sigm(v));
          out[idx] = nv;
          outbf[idx] = (__bf16)nv;   // final launch leaves complete bf16 copy
        } else if constexpr (EPI == 4) {
          outbf[idx] = (__bf16)(0.5f * v * (1.f + erff(v * 0.70710678118654752f)));
        } else {
          out[idx] = v + add1[idx] + rs * resid[idx];
        }
      }
    }
  }
}

__global__ void cvt_bf16_kernel(const float* __restrict__ s, __bf16* __restrict__ d, int n) {
  int i = blockIdx.x * blockDim.x + threadIdx.x;
  if (i < n) d[i] = (__bf16)s[i];
}

__global__ __launch_bounds__(128)
void rmsnorm_kernel(const float* __restrict__ x, const float* __restrict__ w,
                    float* __restrict__ out, __bf16* __restrict__ outbf) {
  int row = blockIdx.x, tid = threadIdx.x;
  const float* xr = x + (size_t)row * DD;
  float vals[4], ss = 0.f;
  #pragma unroll
  for (int i = 0; i < 4; ++i) { float v = xr[tid + i * 128]; vals[i] = v; ss += v * v; }
  for (int off = 16; off; off >>= 1) ss += __shfl_xor(ss, off, 32);
  __shared__ float red[4];
  if ((tid & 31) == 0) red[tid >> 5] = ss;
  __syncthreads();
  float scale = rsqrtf((red[0] + red[1] + red[2] + red[3]) / (float)DD + 1e-6f);
  size_t base = (size_t)row * DD;
  #pragma unroll
  for (int i = 0; i < 4; ++i) {
    int c = tid + i * 128;
    float v = w[c] * vals[i] * scale;
    out[base + c] = v;
    outbf[base + c] = (__bf16)v;
  }
}

__global__ __launch_bounds__(128)
void addrms_kernel(const float* __restrict__ a, const float* __restrict__ b,
                   const float* __restrict__ w, float* __restrict__ out,
                   __bf16* __restrict__ outbf) {
  int row = blockIdx.x, tid = threadIdx.x;
  size_t base = (size_t)row * DD;
  float vals[4], ss = 0.f;
  #pragma unroll
  for (int i = 0; i < 4; ++i) {
    int c = tid + i * 128;
    float v = a[base + c] + b[base + c];
    vals[i] = v; ss += v * v;
  }
  for (int off = 16; off; off >>= 1) ss += __shfl_xor(ss, off, 32);
  __shared__ float red[4];
  if ((tid & 31) == 0) red[tid >> 5] = ss;
  __syncthreads();
  float scale = rsqrtf((red[0] + red[1] + red[2] + red[3]) / (float)DD + 1e-6f);
  #pragma unroll
  for (int i = 0; i < 4; ++i) {
    int c = tid + i * 128;
    float v = w[c] * vals[i] * scale;
    out[base + c] = v;
    outbf[base + c] = (__bf16)v;
  }
}

__global__ void conv_kernel(const float* __restrict__ xn, const float* __restrict__ w,
                            const float* __restrict__ bconv, __bf16* __restrict__ xc) {
  int idx = blockIdx.x * blockDim.x + threadIdx.x;
  if (idx >= NT * DD) return;
  int d = idx % DD;
  int tok = idx / DD;
  int l = tok % LSEQ;
  int b = tok / LSEQ;
  float acc = bconv[d];
  #pragma unroll
  for (int k = 0; k < KCONV; ++k) {
    int ll = l - DIL * (KCONV - 1 - k);
    if (ll >= 0) acc += xn[((size_t)b * LSEQ + ll) * DD + d] * w[d * KCONV + k];
  }
  xc[idx] = (__bf16)acc;
}

__global__ void scan_kernel(const float* __restrict__ v, const float* __restrict__ mem,
                            const float* __restrict__ td, const float* __restrict__ tfirst,
                            __bf16* __restrict__ wsbf, float* __restrict__ next_mem) {
  int t = blockIdx.x * blockDim.x + threadIdx.x;
  if (t >= BBATCH * MM) return;
  int b = t / MM, m = t % MM;
  float decay = 0.9f * sigm(td[m]) + 0.1f;
  float addv = mem[b * MM + m] * sigm(tfirst[m]);
  float s = 0.f;
  size_t base = (size_t)b * LSEQ * MM + m;
  for (int l = 0; l < LSEQ; ++l) {
    size_t p = base + (size_t)l * MM;
    s = s * decay + v[p];
    wsbf[p] = (__bf16)(s + addv);
  }
  next_mem[b * MM + m] = s + addv;   // ws[:, -1]
}

__global__ __launch_bounds__(256)
void router_kernel(const float* __restrict__ h, const float* __restrict__ rw,
                   const float* __restrict__ rb, float* __restrict__ gates,
                   int* __restrict__ gidx) {
  int row = blockIdx.x * 8 + (threadIdx.x >> 5);
  int lane = threadIdx.x & 31;
  const float* hr = h + (size_t)row * DD;
  float a0 = 0, a1 = 0, a2 = 0, a3 = 0;
  for (int j = lane; j < DD; j += 32) {
    float hv = hr[j];
    a0 += hv * rw[0 * DD + j]; a1 += hv * rw[1 * DD + j];
    a2 += hv * rw[2 * DD + j]; a3 += hv * rw[3 * DD + j];
  }
  for (int off = 16; off; off >>= 1) {
    a0 += __shfl_xor(a0, off, 32); a1 += __shfl_xor(a1, off, 32);
    a2 += __shfl_xor(a2, off, 32); a3 += __shfl_xor(a3, off, 32);
  }
  if (lane == 0) {
    float lg[4] = { a0 + rb[0], a1 + rb[1], a2 + rb[2], a3 + rb[3] };
    int i0 = 0;
    for (int e = 1; e < 4; ++e) if (lg[e] > lg[i0]) i0 = e;
    int i1 = -1;
    for (int e = 0; e < 4; ++e) {
      if (e == i0) continue;
      if (i1 < 0 || lg[e] > lg[i1]) i1 = e;
    }
    float e1 = __expf(lg[i1] - lg[i0]);
    float r = fast_rcp(1.f + e1);
    gates[row * 2] = r;
    gates[row * 2 + 1] = e1 * r;
    gidx[row * 2] = i0;
    gidx[row * 2 + 1] = i1;
  }
}

extern "C" void kernel_launch(void* const* d_in, const int* in_sizes, int n_in,
                              void* d_out, int out_size, void* d_ws, size_t ws_size,
                              hipStream_t stream) {
  const float* x          = (const float*)d_in[0];
  const float* mem        = (const float*)d_in[1];
  const float* norm_w     = (const float*)d_in[2];
  const float* dw_w       = (const float*)d_in[3];
  const float* dw_b       = (const float*)d_in[4];
  const float* pw_w       = (const float*)d_in[5];
  const float* pw_b       = (const float*)d_in[6];
  const float* time_decay = (const float*)d_in[7];
  const float* time_first = (const float*)d_in[8];
  const float* v_w        = (const float*)d_in[9];
  const float* v_b        = (const float*)d_in[10];
  const float* gate_w     = (const float*)d_in[11];
  const float* gate_b     = (const float*)d_in[12];
  const float* router_w   = (const float*)d_in[13];
  const float* router_b   = (const float*)d_in[14];
  const float* expert_w   = (const float*)d_in[15];
  const float* expert_b   = (const float*)d_in[16];
  const float* moe_out_w  = (const float*)d_in[17];
  const float* moe_out_b  = (const float*)d_in[18];
  const float* fusion_w   = (const float*)d_in[19];
  const float* ff_down_w  = (const float*)d_in[20];
  const float* ff_down_b  = (const float*)d_in[21];
  const float* ff_up_w    = (const float*)d_in[22];
  const float* ff_up_b    = (const float*)d_in[23];
  const float* rscale     = (const float*)d_in[24];

  char* p = (char*)d_ws;
  float* A    = (float*)p;  p += (size_t)NT * DD * 4;   // xn -> h2 (f32)
  float* C    = (float*)p;  p += (size_t)NT * DD * 4;   // x_conv -> h (f32)
  float* CB   = (float*)p;  p += (size_t)NT * DD * 4;   // combined f32 -> moe f32
  float* V    = (float*)p;  p += (size_t)NT * MM * 4;   // v (f32)
  __bf16* Abf  = (__bf16*)p; p += (size_t)NT * DD * 2;  // xn_bf -> h2_bf
  __bf16* Hbf  = (__bf16*)p; p += (size_t)NT * DD * 2;  // xc_bf -> h_bf
  __bf16* CBbf = (__bf16*)p; p += (size_t)NT * DD * 2;  // combined bf16
  __bf16* Vbf  = (__bf16*)p; p += (size_t)NT * MM * 2;  // ws_bf -> ffh_bf
  float* G  = (float*)p;    p += (size_t)NT * 2 * 4;    // gates
  int*   GI = (int*)p;      p += (size_t)NT * 2 * 4;    // top-2 idx
  __bf16* wbA = (__bf16*)p;                             // bf16 weight arena
  __bf16* pw_wb   = wbA;                 wbA += (size_t)DD * DD;
  __bf16* v_wb    = wbA;                 wbA += (size_t)MM * DD;
  __bf16* gate_wb = wbA;                 wbA += (size_t)DD * (DD + MM);
  __bf16* exp_wb  = wbA;                 wbA += (size_t)EE * DD * DD;
  __bf16* moe_wb  = wbA;                 wbA += (size_t)DD * DD;
  __bf16* ffd_wb  = wbA;                 wbA += (size_t)HH * DD;
  __bf16* ffu_wb  = wbA;                 wbA += (size_t)DD * HH;

  float* out      = (float*)d_out;
  float* next_mem = out + (size_t)NT * DD;

  // 0) one-time (per launch) weight conversions to bf16
  auto cvt = [&](const float* s, __bf16* d, int n) {
    cvt_bf16_kernel<<<(n + 255) / 256, 256, 0, stream>>>(s, d, n);
  };
  cvt(pw_w,      pw_wb,   DD * DD);
  cvt(v_w,       v_wb,    MM * DD);
  cvt(gate_w,    gate_wb, DD * (DD + MM));
  cvt(expert_w,  exp_wb,  EE * DD * DD);
  cvt(moe_out_w, moe_wb,  DD * DD);
  cvt(ff_down_w, ffd_wb,  HH * DD);
  cvt(ff_up_w,   ffu_wb,  DD * HH);

  dim3 blk(128);

  // 1) xn = rmsnorm(x)   (f32 + bf16)
  rmsnorm_kernel<<<NT, 128, 0, stream>>>(x, norm_w, A, Abf);
  // 2) xc = causal dilated depthwise conv(xn)  (bf16)
  conv_kernel<<<(NT * DD + 255) / 256, 256, 0, stream>>>(A, dw_w, dw_b, Hbf);
  // 3) x_conv = silu(xc @ pw_w.T + pw_b)  (f32)
  gemm_wmma<1><<<dim3(DD / 64, NT / 64), blk, 0, stream>>>(
      Hbf, nullptr, DD, DD, pw_wb, pw_b, C, nullptr, DD,
      nullptr, nullptr, nullptr, 0, nullptr, nullptr);
  // 4) v = xn @ v_w.T + v_b  (f32)
  gemm_wmma<0><<<dim3(MM / 64, NT / 64), blk, 0, stream>>>(
      Abf, nullptr, DD, DD, v_wb, v_b, V, nullptr, MM,
      nullptr, nullptr, nullptr, 0, nullptr, nullptr);
  // 5) decay scan -> ws (bf16) + next_memory_state
  scan_kernel<<<(BBATCH * MM + 127) / 128, 128, 0, stream>>>(
      V, mem, time_decay, time_first, Vbf, next_mem);
  // 6) h = x_conv + xn + concat(xn, ws) @ gate_w.T + gate_b  (in place f32 + bf16)
  gemm_wmma<2><<<dim3(DD / 64, NT / 64), blk, 0, stream>>>(
      Abf, Vbf, DD, DD + MM, gate_wb, gate_b, C, Hbf, DD,
      A, nullptr, nullptr, 0, nullptr, nullptr);
  // 7) router top-2 softmax
  router_kernel<<<NT / 8, 256, 0, stream>>>(C, router_w, router_b, G, GI);
  // 8) combined = 0; sequential expert accumulation (stream order => no atomics)
  hipMemsetAsync(CB, 0, (size_t)NT * DD * sizeof(float), stream);
  for (int e = 0; e < EE; ++e) {
    gemm_wmma<3><<<dim3(DD / 64, NT / 64), blk, 0, stream>>>(
        Hbf, nullptr, DD, DD, exp_wb + (size_t)e * DD * DD, expert_b + (size_t)e * DD,
        CB, CBbf, DD, nullptr, G, GI, e, nullptr, nullptr);
  }
  // 9) moe = combined @ moe_out_w.T + moe_out_b  (reads bf16 copy, writes f32 into CB's slot)
  gemm_wmma<0><<<dim3(DD / 64, NT / 64), blk, 0, stream>>>(
      CBbf, nullptr, DD, DD, moe_wb, moe_out_b, CB, nullptr, DD,
      nullptr, nullptr, nullptr, 0, nullptr, nullptr);
  // 10) h2 = rmsnorm(h + moe)  (f32 + bf16)
  addrms_kernel<<<NT, 128, 0, stream>>>(C, CB, fusion_w, A, Abf);
  // 11) ffh = gelu(h2 @ ff_down_w.T + ff_down_b)  (bf16 only)
  gemm_wmma<4><<<dim3(HH / 64, NT / 64), blk, 0, stream>>>(
      Abf, nullptr, DD, DD, ffd_wb, ff_down_b, nullptr, Vbf, HH,
      nullptr, nullptr, nullptr, 0, nullptr, nullptr);
  // 12) out = rs*x + h2 + (ffh @ ff_up_w.T + ff_up_b)
  gemm_wmma<5><<<dim3(DD / 64, NT / 64), blk, 0, stream>>>(
      Vbf, nullptr, HH, HH, ffu_wb, ff_up_b, out, nullptr, DD,
      A, nullptr, nullptr, 0, x, rscale);
}
